// EdgePredictor_65575560675751
// MI455X (gfx1250) — matrix operand order repaired and checked by reading, not compiled, not run on hardware
//
#include <hip/hip_runtime.h>
#include <hip/hip_bf16.h>

typedef __attribute__((ext_vector_type(16))) __bf16 bf16x16;
typedef __attribute__((ext_vector_type(8)))  float  f32x8;

#define D_NODE 128
#define D_EDGE 64
#define D_IN   320
#define HID    256
#define NCLS   8
#define LDS_STRIDE 328   // 16-edge tile row stride in bf16 elems (padded to spread LDS banks)

__device__ __forceinline__ unsigned short f2bf(float f) {
  unsigned int u = __float_as_uint(f);
  u += 0x7FFFu + ((u >> 16) & 1u);   // round-to-nearest-even
  return (unsigned short)(u >> 16);
}

union Frag { uint4 q[2]; bf16x16 v; };

// gfx1250 async copy: LDS[lds_off] <= MEM[saddr + voff], 16 bytes, tracked by ASYNCcnt
__device__ __forceinline__ void async_gather_b128(unsigned lds_off, unsigned voff,
                                                  const unsigned short* base) {
  asm volatile("global_load_async_to_lds_b128 %0, %1, %2 offset:0"
               :: "v"(lds_off), "v"(voff), "s"(base)
               : "memory");
}
__device__ __forceinline__ void wait_asynccnt0() {
  asm volatile("s_wait_asynccnt 0x0" ::: "memory");
}

// ---------------- prep: fp32 -> bf16 conversion / zeroing ----------------
__global__ void cvt_bf16_kernel(const float* __restrict__ in, unsigned short* __restrict__ out, int n4) {
  int i = blockIdx.x * blockDim.x + threadIdx.x;
  if (i >= n4) return;
  float4 v = ((const float4*)in)[i];
  unsigned int p0 = (unsigned int)f2bf(v.x) | ((unsigned int)f2bf(v.y) << 16);
  unsigned int p1 = (unsigned int)f2bf(v.z) | ((unsigned int)f2bf(v.w) << 16);
  ((uint2*)out)[i] = make_uint2(p0, p1);
}

__global__ void zero_kernel(float* __restrict__ p, int n) {
  int i = blockIdx.x * blockDim.x + threadIdx.x;
  if (i < n) p[i] = 0.0f;
}

// ---------------- pass 1: gather + bf16 WMMA GEMM -> channel sum / sumsq ----------------
// Node gathers ride the CDNA5 async-to-LDS path (ASYNCcnt) and overlap the WMMA block
// of the previous tile; edge feats go through registers (need f32->bf16 conversion).
// Double-buffered LDS, one barrier per tile, manually double-buffered A fragments.
__global__ __launch_bounds__(256) void pass1_kernel(
    const float* __restrict__ edge_feats,
    const int*   __restrict__ src_idx,
    const int*   __restrict__ dst_idx,
    const unsigned short* __restrict__ nodeBf,
    const unsigned short* __restrict__ w1Bf,     // [HID][D_IN] bf16
    float* __restrict__ gsum, float* __restrict__ gsumsq,
    int nTiles)
{
  __shared__ unsigned short hbuf[2][16 * LDS_STRIDE];
  const int t    = threadIdx.x;
  const int wave = t >> 5;        // 8 waves, each owns 32 hidden channels
  const int lane = t & 31;
  const int half = lane >> 4;     // 0|1
  const int l15  = lane & 15;

  // Preload B fragments (W1^T): wave's 2 N-tiles x 10 K-steps, resident in VGPRs.
  // B order used here (contiguous k, matches A): lane = column N, lanes 0-15 hold K=0..15, lanes 16-31 K=16..31.
  bf16x16 B[2][10];
  #pragma unroll
  for (int nt = 0; nt < 2; ++nt) {
    const unsigned short* wp = w1Bf + (size_t)(wave * 32 + nt * 16 + l15) * D_IN + half * 16;
    #pragma unroll
    for (int kk = 0; kk < 10; ++kk) {
      Frag f;
      f.q[0] = *(const uint4*)(wp + kk * 32);
      f.q[1] = *(const uint4*)(wp + kk * 32 + 8);
      B[nt][kk] = f.v;
    }
  }

  float s1a = 0.f, s2a = 0.f, s1b = 0.f, s2b = 0.f;

  const int er = t >> 4;           // staging row (edge within tile)
  const int c4 = (t & 15) * 4;     // edge-feat columns (4 floats / thread)
  const int c8 = (t & 15) * 8;     // node-feat columns (8 bf16 / thread)

  // LDS byte offsets (per parity) for the async node writes
  unsigned ldsSrc[2], ldsDst[2];
  #pragma unroll
  for (int p = 0; p < 2; ++p) {
    ldsSrc[p] = (unsigned)(uintptr_t)(&hbuf[p][er * LDS_STRIDE + D_EDGE + c8]);
    ldsDst[p] = (unsigned)(uintptr_t)(&hbuf[p][er * LDS_STRIDE + D_EDGE + D_NODE + c8]);
  }

  float4 pev = make_float4(0.f, 0.f, 0.f, 0.f);
  unsigned nsi = 0, ndi = 0;

  // prologue: prefetch tile0 (block-uniform guard); node rows go straight to LDS buf0
  if (blockIdx.x < nTiles) {
    const long long e0 = (long long)blockIdx.x * 16;
    pev = *(const float4*)(edge_feats + (e0 + er) * D_EDGE + c4);
    nsi = (unsigned)src_idx[e0 + er];
    ndi = (unsigned)dst_idx[e0 + er];
    async_gather_b128(ldsSrc[0], (nsi * D_NODE + c8) * 2u, nodeBf);
    async_gather_b128(ldsDst[0], (ndi * D_NODE + c8) * 2u, nodeBf);
  }

  int parity = 0;
  for (int tile = blockIdx.x; tile < nTiles; tile += gridDim.x) {
    // ---- stage edge part of current tile into LDS buf[parity] ----
    {
      unsigned short* hb = hbuf[parity];
      unsigned int p0 = (unsigned int)f2bf(pev.x) | ((unsigned int)f2bf(pev.y) << 16);
      unsigned int p1 = (unsigned int)f2bf(pev.z) | ((unsigned int)f2bf(pev.w) << 16);
      *(uint2*)(hb + er * LDS_STRIDE + c4) = make_uint2(p0, p1);
    }

    // ---- prefetch next tile's edge floats + indices (block-uniform branch) ----
    const int nextTile = tile + (int)gridDim.x;
    const bool hasNext = nextTile < nTiles;
    if (hasNext) {
      const long long e0 = (long long)nextTile * 16;
      pev = *(const float4*)(edge_feats + (e0 + er) * D_EDGE + c4);
      nsi = (unsigned)src_idx[e0 + er];
      ndi = (unsigned)dst_idx[e0 + er];
    }

    wait_asynccnt0();   // this wave's node rows for 'tile' are in buf[parity]
    __syncthreads();    // + every other wave's async writes & edge ds_stores visible

    // ---- issue async node gathers for next tile into buf[parity^1] ----
    // (post-barrier: all computes of tile-1 from that buffer have completed)
    if (hasNext) {
      async_gather_b128(ldsSrc[parity ^ 1], (nsi * D_NODE + c8) * 2u, nodeBf);
      async_gather_b128(ldsDst[parity ^ 1], (ndi * D_NODE + c8) * 2u, nodeBf);
    }

    // ---- WMMA: A from LDS (double-buffered fragments), resident B ----
    // A order used here (contiguous k, matches B): lane row = l15; lanes 0-15: K 0..15, lanes 16-31: K 16..31
    f32x8 acc0 = {}; f32x8 acc1 = {};
    const unsigned short* ap = hbuf[parity] + l15 * LDS_STRIDE + half * 16;
    Frag fa[2];
    fa[0].q[0] = *(const uint4*)(ap + 0);
    fa[0].q[1] = *(const uint4*)(ap + 8);
    #pragma unroll
    for (int kk = 0; kk < 10; ++kk) {
      const int cur = kk & 1;
      if (kk < 9) {
        fa[cur ^ 1].q[0] = *(const uint4*)(ap + (kk + 1) * 32);
        fa[cur ^ 1].q[1] = *(const uint4*)(ap + (kk + 1) * 32 + 8);
      }
      acc0 = __builtin_amdgcn_wmma_f32_16x16x32_bf16(false, fa[cur].v, false, B[0][kk], (short)0, acc0, false, false);
      acc1 = __builtin_amdgcn_wmma_f32_16x16x32_bf16(false, fa[cur].v, false, B[1][kk], (short)0, acc1, false, false);
    }
    #pragma unroll
    for (int r = 0; r < 8; ++r) {
      s1a += acc0[r]; s2a += acc0[r] * acc0[r];
      s1b += acc1[r]; s2b += acc1[r] * acc1[r];
    }
    parity ^= 1;
  }

  // Combine lane halves (rows M 0-7 with 8-15 of same column), then one atomic per channel.
  s1a += __shfl_xor(s1a, 16, 32);
  s2a += __shfl_xor(s2a, 16, 32);
  s1b += __shfl_xor(s1b, 16, 32);
  s2b += __shfl_xor(s2b, 16, 32);
  if (lane < 16) {
    atomicAdd(&gsum  [wave * 32 + lane],      s1a);
    atomicAdd(&gsumsq[wave * 32 + lane],      s2a);
    atomicAdd(&gsum  [wave * 32 + 16 + lane], s1b);
    atomicAdd(&gsumsq[wave * 32 + 16 + lane], s2b);
  }
}

// ---------------- finalize: fold BN + Linear2 into M[8x320], cbias[8] ----------------
__global__ __launch_bounds__(256) void finalize_kernel(
    const float* __restrict__ gsum, const float* __restrict__ gsumsq,
    const float* __restrict__ gamma, const float* __restrict__ beta,
    const float* __restrict__ W1, const float* __restrict__ W2,
    const float* __restrict__ b2, float Ef,
    float* __restrict__ Mfold, float* __restrict__ cbias)
{
  __shared__ float a_s[HID], d_s[HID];
  int t = threadIdx.x;
  {
    float invE = 1.0f / Ef;
    float mu  = gsum[t] * invE;
    float var = gsumsq[t] * invE - mu * mu;
    float sc  = rsqrtf(var + 1e-5f) * gamma[t];
    a_s[t] = sc;
    d_s[t] = beta[t] - mu * sc;
  }
  __syncthreads();
  for (int o = t; o < NCLS * D_IN; o += 256) {
    int c = o / D_IN, k = o - c * D_IN;
    float s = 0.f;
    for (int h = 0; h < HID; ++h)
      s = fmaf(W2[c * HID + h] * a_s[h], W1[h * D_IN + k], s);
    Mfold[o] = s;
  }
  if (t < NCLS) {
    float s = b2[t];
    for (int h = 0; h < HID; ++h)
      s = fmaf(d_s[h], W2[t * HID + h], s);
    cbias[t] = s;
  }
}

// ---------------- pass 2: logits = h . M^T + cbias, log_softmax ----------------
__global__ __launch_bounds__(256) void pass2_kernel(
    const float* __restrict__ edge_feats, const float* __restrict__ n_feats,
    const int* __restrict__ src_idx, const int* __restrict__ dst_idx,
    const float* __restrict__ Mfold, const float* __restrict__ cbias,
    float* __restrict__ out, int E)
{
  int e = blockIdx.x * 256 + threadIdx.x;
  if (e >= E) return;

  float acc[NCLS];
  #pragma unroll
  for (int c = 0; c < NCLS; ++c) acc[c] = cbias[c];   // uniform -> scalar loads

  const float4* ef = (const float4*)(edge_feats + (long long)e * D_EDGE);
  #pragma unroll 4
  for (int k4 = 0; k4 < D_EDGE / 4; ++k4) {
    float4 v = ef[k4];
    #pragma unroll
    for (int c = 0; c < NCLS; ++c) {
      const float* m = Mfold + c * D_IN + k4 * 4;     // uniform address -> SGPR operand
      acc[c] = fmaf(v.x, m[0], acc[c]);
      acc[c] = fmaf(v.y, m[1], acc[c]);
      acc[c] = fmaf(v.z, m[2], acc[c]);
      acc[c] = fmaf(v.w, m[3], acc[c]);
    }
  }
  const float4* sf = (const float4*)(n_feats + (long long)src_idx[e] * D_NODE);
  #pragma unroll 4
  for (int k4 = 0; k4 < D_NODE / 4; ++k4) {
    float4 v = sf[k4];
    #pragma unroll
    for (int c = 0; c < NCLS; ++c) {
      const float* m = Mfold + c * D_IN + D_EDGE + k4 * 4;
      acc[c] = fmaf(v.x, m[0], acc[c]);
      acc[c] = fmaf(v.y, m[1], acc[c]);
      acc[c] = fmaf(v.z, m[2], acc[c]);
      acc[c] = fmaf(v.w, m[3], acc[c]);
    }
  }
  const float4* df = (const float4*)(n_feats + (long long)dst_idx[e] * D_NODE);
  #pragma unroll 4
  for (int k4 = 0; k4 < D_NODE / 4; ++k4) {
    float4 v = df[k4];
    #pragma unroll
    for (int c = 0; c < NCLS; ++c) {
      const float* m = Mfold + c * D_IN + D_EDGE + D_NODE + k4 * 4;
      acc[c] = fmaf(v.x, m[0], acc[c]);
      acc[c] = fmaf(v.y, m[1], acc[c]);
      acc[c] = fmaf(v.z, m[2], acc[c]);
      acc[c] = fmaf(v.w, m[3], acc[c]);
    }
  }

  float mx = acc[0];
  #pragma unroll
  for (int c = 1; c < NCLS; ++c) mx = fmaxf(mx, acc[c]);
  float s = 0.f;
  #pragma unroll
  for (int c = 0; c < NCLS; ++c) s += __expf(acc[c] - mx);
  float lse = mx + __logf(s);

  float4* op = (float4*)(out + (long long)e * NCLS);
  op[0] = make_float4(acc[0] - lse, acc[1] - lse, acc[2] - lse, acc[3] - lse);
  op[1] = make_float4(acc[4] - lse, acc[5] - lse, acc[6] - lse, acc[7] - lse);
}

// ---------------- host ----------------
extern "C" void kernel_launch(void* const* d_in, const int* in_sizes, int n_in,
                              void* d_out, int out_size, void* d_ws, size_t ws_size,
                              hipStream_t stream) {
  const float* n_feats    = (const float*)d_in[0];
  const float* edge_feats = (const float*)d_in[1];
  const int*   src_idx    = (const int*)d_in[2];
  const int*   dst_idx    = (const int*)d_in[3];
  const float* W1         = (const float*)d_in[4];
  // d_in[5] = b1 : cancels exactly in training-mode BatchNorm, unused
  const float* gamma      = (const float*)d_in[6];
  const float* beta       = (const float*)d_in[7];
  const float* W2         = (const float*)d_in[8];
  const float* b2         = (const float*)d_in[9];
  float* out = (float*)d_out;

  const int E      = in_sizes[2];
  const int nNodes = in_sizes[0] / D_NODE;

  char* ws = (char*)d_ws;
  unsigned short* w1Bf   = (unsigned short*)ws;                                  // 81920 bf16
  unsigned short* nodeBf = (unsigned short*)(ws + (size_t)HID * D_IN * 2);       // nNodes*128 bf16
  size_t off = (size_t)HID * D_IN * 2 + (size_t)nNodes * D_NODE * 2;
  off = (off + 255) & ~(size_t)255;
  float* gsum   = (float*)(ws + off);        // 256
  float* gsumsq = gsum + HID;                // 256
  float* Mfold  = gsumsq + HID;              // 2560
  float* cbias  = Mfold + NCLS * D_IN;       // 8

  // prep
  zero_kernel<<<2, 256, 0, stream>>>(gsum, 2 * HID);
  cvt_bf16_kernel<<<(HID * D_IN / 4 + 255) / 256, 256, 0, stream>>>(W1, w1Bf, HID * D_IN / 4);
  cvt_bf16_kernel<<<(nNodes * D_NODE / 4 + 255) / 256, 256, 0, stream>>>(n_feats, nodeBf, nNodes * D_NODE / 4);

  // pass 1: WMMA GEMM + BN statistics (pipelined, async node gathers into LDS)
  const int nTiles = E / 16;  // E = 800000, divisible
  pass1_kernel<<<1250, 256, 0, stream>>>(edge_feats, src_idx, dst_idx, nodeBf, w1Bf, gsum, gsumsq, nTiles);

  // fold BN + W2 into 8x320 matrix
  finalize_kernel<<<1, 256, 0, stream>>>(gsum, gsumsq, gamma, beta, W1, W2, b2, (float)E, Mfold, cbias);

  // pass 2: logits + log_softmax
  pass2_kernel<<<(E + 255) / 256, 256, 0, stream>>>(edge_feats, n_feats, src_idx, dst_idx, Mfold, cbias, out, E);
}
